// SpikingJellyModel_86474871538259
// MI455X (gfx1250) — compile-verified, matrix operand order checked by reading, not run on hardware
//
#include <hip/hip_runtime.h>
#include <cstdint>
#include <cstddef>

// Problem constants (match reference)
#define B_SZ    500
#define T_SZ    100
#define I_SZ    700
#define H_SZ    512
#define O_SZ    20
#define HALF_I  350
#define KPAD    352            // 350 padded to multiple of 32
#define ROWS    50000          // T*B rows, row index r = t*B + b
#define ROWPITCH 704           // 2*KPAD halfs per inph row

typedef _Float16 half_t;
typedef __attribute__((ext_vector_type(16))) _Float16 v16h;
typedef __attribute__((ext_vector_type(8)))  _Float16 v8h;
typedef __attribute__((ext_vector_type(8)))  float    v8f;
typedef int i32x4_t __attribute__((vector_size(16)));

union FragU { v16h v; struct { v8h lo; v8h hi; } p; };

// Async global->LDS copy support (gfx1250). Guarded so absence of the
// builtin falls back to the sync staging path.
#if defined(__has_builtin)
#  if __has_builtin(__builtin_amdgcn_global_load_async_to_lds_b128)
#    define HAVE_ASYNC_LDS 1
#  endif
#endif
#ifndef HAVE_ASYNC_LDS
#  define HAVE_ASYNC_LDS 0
#endif

#if HAVE_ASYNC_LDS
// Prototype (from clang diagnostic): param0 = int32x4 addrspace(1)*,
// param1 = int32x4 addrspace(3)*, then imm offset, imm cpol.
#define ASYNC_CP16(gsrc, ldst)                                               \
    __builtin_amdgcn_global_load_async_to_lds_b128(                          \
        (__attribute__((address_space(1))) i32x4_t*)(gsrc),                  \
        (__attribute__((address_space(3))) i32x4_t*)(ldst), 0, 0)
static __device__ __forceinline__ void wait_async0() {
#if __has_builtin(__builtin_amdgcn_s_wait_asynccnt)
    __builtin_amdgcn_s_wait_asynccnt(0);
#else
    asm volatile("s_wait_asynccnt 0" ::: "memory");
#endif
}
#endif

// ---------------------------------------------------------------------------
// Phase 0: init output scalar slots (loss, correct, samples)
// ---------------------------------------------------------------------------
__global__ void init_out_kernel(float* __restrict__ out) {
    if (threadIdx.x == 0 && blockIdx.x == 0) {
        out[0] = 0.0f;                       // total_loss
        out[1 + 1000000] = 0.0f;             // total_correct
        int cnt = 0;
        for (int t = 0; t < T_SZ; ++t)
            if (((t - 10) % 15 > 5) && (t > 10)) cnt++;
        out[1 + 1000000 + 1] = (float)(cnt * B_SZ);   // total_samples
    }
}

// ---------------------------------------------------------------------------
// Phase 1a: weights fp32 -> f16, K-padded.  w1h/w2h: [512][352]; wouth: [32][512]
// ---------------------------------------------------------------------------
__global__ void prep_weights_kernel(const float* __restrict__ W1,
                                    const float* __restrict__ W2,
                                    const float* __restrict__ Wout,
                                    half_t* __restrict__ w1h,
                                    half_t* __restrict__ w2h,
                                    half_t* __restrict__ wouth) {
    int idx = blockIdx.x * blockDim.x + threadIdx.x;
    const int NW = H_SZ * KPAD;
    if (idx < 2 * NW) {
        int sel = idx >= NW;
        int e = idx - sel * NW;
        int h = e / KPAD, k = e - h * KPAD;
        float v = (k < HALF_I) ? (sel ? W2[h * HALF_I + k] : W1[h * HALF_I + k]) : 0.0f;
        (sel ? w2h : w1h)[e] = (half_t)v;
    } else {
        int e = idx - 2 * NW;
        if (e < 32 * H_SZ) {
            int n = e >> 9, k = e & (H_SZ - 1);
            wouth[e] = (half_t)((n < O_SZ) ? Wout[n * H_SZ + k] : 0.0f);
        }
    }
}

// ---------------------------------------------------------------------------
// Phase 1b: input fp32 [B,T,I] -> f16 [t*B+b][2][KPAD] (transposed, split halves)
// ---------------------------------------------------------------------------
__global__ void prep_input_kernel(const float* __restrict__ in,
                                  half_t* __restrict__ inph) {
    const long n = (long)B_SZ * T_SZ * I_SZ;
    for (long o = (long)blockIdx.x * blockDim.x + threadIdx.x; o < n;
         o += (long)gridDim.x * blockDim.x) {
        long bt = o / I_SZ;
        int  i  = (int)(o - bt * I_SZ);
        int  b  = (int)(bt / T_SZ);
        int  t  = (int)(bt - (long)b * T_SZ);
        int  slot = (i >= HALF_I) ? 1 : 0;
        int  pos  = i - slot * HALF_I;
        inph[((long)t * B_SZ + b) * ROWPITCH + slot * KPAD + pos] = (half_t)in[o];
    }
}

__global__ void prep_pad_kernel(half_t* __restrict__ inph) {
    int idx = blockIdx.x * blockDim.x + threadIdx.x;
    if (idx >= ROWS * 4) return;
    int row = idx >> 2;
    int q   = idx & 3;
    inph[(long)row * ROWPITCH + (q >> 1) * KPAD + HALF_I + (q & 1)] = (half_t)0.0f;
}

// ---------------------------------------------------------------------------
// Phase 2: drive GEMMs via WMMA f16->f32.
// C[r][h] = sum_k inph[r][branch][k] * W[h][k],  C layout [T,B,H]
// Block: 256 thr = 8 waves (4Mx2N), block tile 128x128, wave tile 32x64.
// ---------------------------------------------------------------------------
#define BM 128
#define BN 128
// Padded LDS row pitch (halfs): 56 halfs = 112B = 7*16B -> every b128 LDS
// access is 16B aligned, and the 16 fragment rows (row*28 banks mod 64)
// land on 16 distinct 4-bank groups -> conflict-free ds_load_b128.
#define LDT 56

static __device__ __forceinline__ void compute_tile(
    const half_t* __restrict__ As, const half_t* __restrict__ Bs,
    int wm, int wn, int l15, int koff, v8f acc[2][4]) {
    FragU af[2], bf[4];
#pragma unroll
    for (int i = 0; i < 2; ++i) {
        const half_t* p = &As[(wm * 32 + i * 16 + l15) * LDT + koff];
        af[i].p.lo = *(const v8h*)p;
        af[i].p.hi = *(const v8h*)(p + 16);
    }
#pragma unroll
    for (int j = 0; j < 4; ++j) {
        const half_t* p = &Bs[(wn * 64 + j * 16 + l15) * LDT + koff];
        bf[j].p.lo = *(const v8h*)p;
        bf[j].p.hi = *(const v8h*)(p + 16);
    }
#pragma unroll
    for (int i = 0; i < 2; ++i)
#pragma unroll
        for (int j = 0; j < 4; ++j)
            acc[i][j] = __builtin_amdgcn_wmma_f32_16x16x32_f16(
                false, af[i].v, false, bf[j].v, (short)0, acc[i][j],
                false, false);
}

__global__ __launch_bounds__(256) void gemm_drive_kernel(
    const half_t* __restrict__ inph,
    const half_t* __restrict__ w1h,
    const half_t* __restrict__ w2h,
    float* __restrict__ c1,
    float* __restrict__ c2) {
    const int branch = blockIdx.z;
    const half_t* __restrict__ W = branch ? w2h : w1h;
    float* __restrict__ C = branch ? c2 : c1;
    const int rowBlk = blockIdx.x * BM;
    const int colBlk = blockIdx.y * BN;

    const int tid  = threadIdx.x;
    const int lane = tid & 31;
    const int wave = tid >> 5;
    const int wm   = wave >> 1;     // 0..3
    const int wn   = wave & 1;      // 0..1
    const int l15  = lane & 15;
    const int koff = (lane >> 4) * 8;

    v8f acc[2][4];
#pragma unroll
    for (int i = 0; i < 2; ++i)
#pragma unroll
        for (int j = 0; j < 4; ++j) acc[i][j] = (v8f){};

    // Staging map: thread -> (row, 16-half segment)
    const int srow = tid >> 1;
    const int sseg = (tid & 1) * 16;
    const long rA  = (long)rowBlk + srow;
    const bool aValid = (rA < ROWS);
    const half_t* aSrc = inph + rA * ROWPITCH + branch * KPAD;
    const half_t* bSrc = W + (long)(colBlk + srow) * KPAD;

#if HAVE_ASYNC_LDS
    // Double-buffered async global->LDS pipeline (ASYNCcnt tracked).
    __shared__ half_t As[2][BM * LDT];
    __shared__ half_t Bs[2][BN * LDT];

    auto stage = [&](int k0, int p) {
        half_t* Ad = &As[p][srow * LDT + sseg];
        half_t* Bd = &Bs[p][srow * LDT + sseg];
        if (aValid) {
            ASYNC_CP16(aSrc + k0 + sseg,     Ad);
            ASYNC_CP16(aSrc + k0 + sseg + 8, Ad + 8);
        }
        ASYNC_CP16(bSrc + k0 + sseg,     Bd);
        ASYNC_CP16(bSrc + k0 + sseg + 8, Bd + 8);
    };

    stage(0, 0);
    const int NK = KPAD / 32;   // 11
    for (int ki = 0; ki < NK; ++ki) {
        const int p = ki & 1;
        wait_async0();          // my copies into buffer p have landed
        __syncthreads();        // everyone's copies landed; prev reads done
        if (ki + 1 < NK) stage((ki + 1) * 32, p ^ 1);
        compute_tile(As[p], Bs[p], wm, wn, l15, koff, acc);
    }
#else
    // Fallback: sync staging via VGPRs, single buffer.
    __shared__ half_t As[1][BM * LDT];
    __shared__ half_t Bs[1][BN * LDT];
    for (int k0 = 0; k0 < KPAD; k0 += 32) {
        if (k0 + 32 < KPAD) {
            __builtin_prefetch(aSrc + k0 + 32, 0, 0);
            __builtin_prefetch(bSrc + k0 + 32, 0, 0);
        }
        v8h a0 = (v8h){}, a1 = (v8h){};
        if (aValid) {
            a0 = *(const v8h*)(aSrc + k0 + sseg);
            a1 = *(const v8h*)(aSrc + k0 + sseg + 8);
        }
        *(v8h*)(&As[0][srow * LDT + sseg])     = a0;
        *(v8h*)(&As[0][srow * LDT + sseg + 8]) = a1;
        v8h b0 = *(const v8h*)(bSrc + k0 + sseg);
        v8h b1 = *(const v8h*)(bSrc + k0 + sseg + 8);
        *(v8h*)(&Bs[0][srow * LDT + sseg])     = b0;
        *(v8h*)(&Bs[0][srow * LDT + sseg + 8]) = b1;
        __syncthreads();
        compute_tile(As[0], Bs[0], wm, wn, l15, koff, acc);
        __syncthreads();
    }
#endif

    // D layout: lane<16 -> n=lane, m=vgpr; lane>=16 -> n=lane-16, m=8+vgpr
#pragma unroll
    for (int i = 0; i < 2; ++i) {
        const long mBase = (long)rowBlk + wm * 32 + i * 16 + (lane >> 4) * 8;
#pragma unroll
        for (int j = 0; j < 4; ++j) {
            const int col = colBlk + wn * 64 + j * 16 + l15;
#pragma unroll
            for (int m = 0; m < 8; ++m) {
                long r = mBase + m;
                if (r < ROWS) C[r * H_SZ + col] = acc[i][j][m];
            }
        }
    }
}

// ---------------------------------------------------------------------------
// Phase 3: sequential LIF scan over T; one thread per (b,h).
// ---------------------------------------------------------------------------
__global__ void scan_kernel(const float* __restrict__ c1,
                            const float* __restrict__ c2,
                            const float* __restrict__ v_init,
                            const float* __restrict__ tau_n,
                            const float* __restrict__ tau_m,
                            half_t* __restrict__ spikes) {
    int j = blockIdx.x * blockDim.x + threadIdx.x;
    if (j >= B_SZ * H_SZ) return;
    int b = j / H_SZ, h = j & (H_SZ - 1);
    float beta1 = 1.0f / (1.0f + __expf(-tau_n[h]));
    float beta2 = 1.0f / (1.0f + __expf(-tau_n[H_SZ + h]));
    float alpha = 1.0f / (1.0f + __expf(-tau_m[h]));
    float ob1 = 1.0f - beta1, ob2 = 1.0f - beta2, oa = 1.0f - alpha;
    float v = v_init[j];
    float d1 = 0.0f, d2 = 0.0f, s = 0.0f;
    for (int t = 0; t < T_SZ; ++t) {
        long idx = ((long)t * B_SZ + b) * H_SZ + h;
        float x1 = c1[idx], x2 = c2[idx];
        d1 = beta1 * d1 + ob1 * x1;
        d2 = beta2 * d2 + ob2 * x2;
        v  = alpha * v + oa * (d1 + d2) - s;     // V_TH = 1
        s  = (v - 1.0f) > 0.0f ? 1.0f : 0.0f;
        spikes[idx] = (half_t)s;
    }
}

// ---------------------------------------------------------------------------
// Phase 4: output GEMM via WMMA + bias.  out[b][t][o], o<20, N padded to 32.
// One wave owns 4 row-tiles (64 rows) so Wout fragments are reused 4x.
// All tile guards are wave-uniform -> EXEC all-ones at every WMMA.
// ---------------------------------------------------------------------------
__global__ __launch_bounds__(256) void gemm_out_kernel(
    const half_t* __restrict__ spikes,   // [ROWS][512]
    const half_t* __restrict__ wouth,    // [32][512]
    const float* __restrict__ bout,      // [20]
    float* __restrict__ out) {           // d_out+1, [B][T][20]
    const int lane = threadIdx.x & 31;
    const int wave = threadIdx.x >> 5;
    const long base = (long)blockIdx.x * 512 + wave * 64;
    if (base >= ROWS) return;
    const int l15  = lane & 15;
    const int koff = (lane >> 4) * 8;

    v8f acc[4][2];
#pragma unroll
    for (int t = 0; t < 4; ++t) { acc[t][0] = (v8f){}; acc[t][1] = (v8f){}; }

    bool valid[4];
    const half_t* aRow[4];
#pragma unroll
    for (int t = 0; t < 4; ++t) {
        valid[t] = (base + t * 16) < ROWS;      // wave-uniform
        aRow[t]  = spikes + (base + t * 16 + l15) * H_SZ;
    }

    for (int k0 = 0; k0 < H_SZ; k0 += 32) {
        FragU bf[2];
#pragma unroll
        for (int j = 0; j < 2; ++j) {
            const half_t* p = wouth + (j * 16 + l15) * H_SZ + k0 + koff;
            bf[j].p.lo = *(const v8h*)p;
            bf[j].p.hi = *(const v8h*)(p + 16);
        }
#pragma unroll
        for (int t = 0; t < 4; ++t) {
            if (!valid[t]) continue;
            FragU af;
            af.p.lo = *(const v8h*)(aRow[t] + k0 + koff);
            af.p.hi = *(const v8h*)(aRow[t] + k0 + koff + 16);
#pragma unroll
            for (int j = 0; j < 2; ++j)
                acc[t][j] = __builtin_amdgcn_wmma_f32_16x16x32_f16(
                    false, af.v, false, bf[j].v, (short)0, acc[t][j],
                    false, false);
        }
    }

#pragma unroll
    for (int t = 0; t < 4; ++t) {
        if (!valid[t]) continue;
#pragma unroll
        for (int j = 0; j < 2; ++j) {
            int n = j * 16 + l15;
            if (n < O_SZ) {
                float bias = bout[n];
#pragma unroll
                for (int m = 0; m < 8; ++m) {
                    long r = base + t * 16 + (lane >> 4) * 8 + m;  // r = t*B + b
                    int tt = (int)(r / B_SZ);
                    int bb = (int)(r - (long)tt * B_SZ);
                    out[((long)bb * T_SZ + tt) * O_SZ + n] = acc[t][j][m] + bias;
                }
            }
        }
    }
}

// ---------------------------------------------------------------------------
// Phase 5: masked double-softmax CE + argmax, atomics into scalar slots.
// ---------------------------------------------------------------------------
__global__ void loss_kernel(const float* __restrict__ outHist,  // d_out+1
                            const int* __restrict__ target,     // [B,T]
                            float* __restrict__ loss,
                            float* __restrict__ correct) {
    int j = blockIdx.x * blockDim.x + threadIdx.x;
    if (j >= B_SZ * T_SZ) return;
    int t = j % T_SZ;
    bool mask = ((t - 10) % 15 > 5) && (t > 10);
    if (!mask) return;
    const float* z = outHist + (long)j * O_SZ;
    float zr[O_SZ];
    float m = -1e30f;
#pragma unroll
    for (int o = 0; o < O_SZ; ++o) { zr[o] = z[o]; m = fmaxf(m, zr[o]); }
    float sum = 0.0f;
    float p[O_SZ];
#pragma unroll
    for (int o = 0; o < O_SZ; ++o) { p[o] = __expf(zr[o] - m); sum += p[o]; }
    float inv = 1.0f / sum;
    float s2 = 0.0f;
#pragma unroll
    for (int o = 0; o < O_SZ; ++o) { p[o] *= inv; s2 += __expf(p[o]); }
    float lse = __logf(s2);
    int tgt = target[j];
    float ce = lse - p[tgt];                 // -(p[tgt] - logsumexp(p))
    int am = 0;
    float best = zr[0];
#pragma unroll
    for (int o = 1; o < O_SZ; ++o)
        if (zr[o] > best) { best = zr[o]; am = o; }
    atomicAdd(loss, ce * (1.0f / (float)B_SZ));
    if (am == tgt) atomicAdd(correct, 1.0f);
}

// ---------------------------------------------------------------------------
extern "C" void kernel_launch(void* const* d_in, const int* in_sizes, int n_in,
                              void* d_out, int out_size, void* d_ws, size_t ws_size,
                              hipStream_t stream) {
    const float* input  = (const float*)d_in[0];
    const int*   target = (const int*)  d_in[1];
    const float* v_init = (const float*)d_in[2];
    const float* W1     = (const float*)d_in[3];
    const float* W2     = (const float*)d_in[4];
    const float* tau_n  = (const float*)d_in[5];
    const float* tau_m  = (const float*)d_in[6];
    const float* Wout   = (const float*)d_in[7];
    const float* bout   = (const float*)d_in[8];
    float* out = (float*)d_out;

    char* ws = (char*)d_ws;
    size_t off = 0;
    auto take = [&](size_t bytes) -> char* {
        char* p = ws + off;
        off = (off + bytes + 255) & ~(size_t)255;
        return p;
    };
    half_t* inph  = (half_t*)take((size_t)ROWS * ROWPITCH * sizeof(half_t));
    half_t* w1h   = (half_t*)take((size_t)H_SZ * KPAD * sizeof(half_t));
    half_t* w2h   = (half_t*)take((size_t)H_SZ * KPAD * sizeof(half_t));
    half_t* wouth = (half_t*)take((size_t)32 * H_SZ * sizeof(half_t));
    float*  c1    = (float*) take((size_t)ROWS * H_SZ * sizeof(float));
    float*  c2    = (float*) take((size_t)ROWS * H_SZ * sizeof(float));
    half_t* spk   = (half_t*)take((size_t)ROWS * H_SZ * sizeof(half_t));
    (void)ws_size; (void)in_sizes; (void)n_in; (void)out_size;

    init_out_kernel<<<1, 1, 0, stream>>>(out);

    int prepW = (2 * H_SZ * KPAD + 32 * H_SZ + 255) / 256;
    prep_weights_kernel<<<prepW, 256, 0, stream>>>(W1, W2, Wout, w1h, w2h, wouth);
    prep_input_kernel<<<2048, 256, 0, stream>>>(input, inph);
    prep_pad_kernel<<<(ROWS * 4 + 255) / 256, 256, 0, stream>>>(inph);

    dim3 gDrive((ROWS + BM - 1) / BM, H_SZ / BN, 2);
    gemm_drive_kernel<<<gDrive, 256, 0, stream>>>(inph, w1h, w2h, c1, c2);

    scan_kernel<<<(B_SZ * H_SZ + 255) / 256, 256, 0, stream>>>(c1, c2, v_init,
                                                               tau_n, tau_m, spk);

    gemm_out_kernel<<<(ROWS + 511) / 512, 256, 0, stream>>>(spk, wouth, bout,
                                                            out + 1);

    loss_kernel<<<(B_SZ * T_SZ + 255) / 256, 256, 0, stream>>>(
        out + 1, target, out, out + 1 + 1000000);
}
